// MaskEstimationModule_49331994362217
// MI455X (gfx1250) — compile-verified, matrix operand order checked by reading, not run on hardware
//
#include <hip/hip_runtime.h>
#include <hip/hip_bf16.h>
#include <stdint.h>

// ---------------- types for WMMA ----------------
typedef __attribute__((ext_vector_type(16))) __bf16 v16bf;
typedef __attribute__((ext_vector_type(8)))  float  v8f;

union Frag { uint4 q[2]; v16bf v; };

__device__ __forceinline__ unsigned short f2bf(float f) {
  unsigned u = __float_as_uint(f);
  unsigned r = u + 0x7FFFu + ((u >> 16) & 1u);   // RNE
  return (unsigned short)(r >> 16);
}
__device__ __forceinline__ float bf2f(unsigned short h) {
  return __uint_as_float(((unsigned)h) << 16);
}

// ---------------- core WMMA 16(M)x64(N) tile per wave (bf16 -> f32) ----------
// A: [M, lda] bf16 row-major (activations). Bt: [NpadRows, ldb] bf16 row-major
// (transposed weights, Bt[n][k] = B[k][n]).  16-bit A-matrix K striping per
// ISA: lanes 0-15 hold K groups {0-7,16-23}, lanes 16-31 hold {8-15,24-31};
// each group is one aligned 16-byte load.  A fragment is reused across the 4
// N-subtiles (4 WMMAs per K-step).
__device__ __forceinline__ void wmma_tile4(const unsigned short* A,
                                           const unsigned short* Bt,
                                           int Kd, int lda, int ldb,
                                           int mbase, int mClamp,
                                           int nbase, int NpadRows,
                                           int lane, v8f acc[4]) {
  int lm = lane & 15;
  int kh = lane >> 4;
  int am = mbase + lm; if (am > mClamp) am = mClamp;   // clamp (stores masked)
  const unsigned short* arow = A + (size_t)am * lda + kh * 8;
  const unsigned short* brow[4];
#pragma unroll
  for (int j = 0; j < 4; ++j) {
    int bn = nbase + j * 16 + lm;
    if (bn > NpadRows - 1) bn = NpadRows - 1;          // clamp into padded Bt
    brow[j] = Bt + (size_t)bn * ldb + kh * 8;
  }
  for (int k0 = 0; k0 < Kd; k0 += 32) {
    Frag fa;
    fa.q[0] = *reinterpret_cast<const uint4*>(arow + k0);
    fa.q[1] = *reinterpret_cast<const uint4*>(arow + k0 + 16);
#pragma unroll
    for (int j = 0; j < 4; ++j) {
      Frag fb;
      fb.q[0] = *reinterpret_cast<const uint4*>(brow[j] + k0);
      fb.q[1] = *reinterpret_cast<const uint4*>(brow[j] + k0 + 16);
      acc[j] = __builtin_amdgcn_wmma_f32_16x16x32_bf16(
          false, fa.v, false, fb.v, (short)0, acc[j], false, false);
    }
  }
}

// D layout: VGPR q -> M = q + 8*(lane>>4), N = lane&15
__device__ __forceinline__ void store_tile(v8f acc, unsigned short* Out,
                                           int ldout, int mbase, int M,
                                           int ncol, int Nreal, float bias,
                                           bool do_tanh, int lane) {
  if (ncol >= Nreal) return;
  int kh = lane >> 4;
#pragma unroll
  for (int q = 0; q < 8; ++q) {
    int m = mbase + q + 8 * kh;
    if (m < M) {
      float v = acc[q] + bias;
      if (do_tanh) v = tanhf(v);
      Out[(size_t)m * ldout + ncol] = f2bf(v);
    }
  }
}

// ---------------- kernels ----------------
#define T_DIM 517
#define FC_DIM 128
#define HID_DIM 512
#define NBANDS 41
#define TN 66176      // 517*128
#define HN 264704     // 517*512

__global__ void layernorm_kernel(const float* __restrict__ x,
                                 unsigned short* __restrict__ XN) {
  __shared__ float ssum[256], ssq[256];
  const int tid = threadIdx.x;
  const float* xb = x + (size_t)blockIdx.x * TN;
  float s = 0.f, sq = 0.f;
  for (int i = tid; i < TN; i += 256) { float v = xb[i]; s += v; sq += v * v; }
  ssum[tid] = s; ssq[tid] = sq;
  __syncthreads();
  for (int o = 128; o > 0; o >>= 1) {
    if (tid < o) { ssum[tid] += ssum[tid + o]; ssq[tid] += ssq[tid + o]; }
    __syncthreads();
  }
  float mean = ssum[0] * (1.f / TN);
  float var  = ssq[0] * (1.f / TN) - mean * mean;
  float rs   = rsqrtf(var + 1e-5f);
  unsigned short* ob = XN + (size_t)blockIdx.x * TN;
  for (int i = tid; i < TN; i += 256) ob[i] = f2bf((xb[i] - mean) * rs);
}

// dst[n*Kpad + c] = (n<Nsrc && c<Ksrc) ? src[c*Nsrc + n] : 0  (bf16 transpose-pack)
__global__ void pack_bt_kernel(const float* __restrict__ src,
                               unsigned short* __restrict__ dst,
                               int Nsrc, int Ksrc, int Npad, int Kpad) {
  int idx = blockIdx.x * 256 + threadIdx.x;
  int tot = Npad * Kpad;
  if (idx >= tot) return;
  int n = idx / Kpad, c = idx % Kpad;
  float v = (n < Nsrc && c < Ksrc) ? src[(size_t)c * Nsrc + n] : 0.f;
  dst[idx] = f2bf(v);
}

__global__ void zero_u32_kernel(unsigned int* __restrict__ p, long long n) {
  long long i = (long long)blockIdx.x * 256 + threadIdx.x;
  if (i < n) p[i] = 0u;
}

// GEMM1: h = tanh(xn[z] @ w1[k] + b1[k]), z = b*41+k
// block = 4 waves over M; each wave: 16 rows x 64 cols
__global__ void gemm1_kernel(const unsigned short* __restrict__ XN,
                             const unsigned short* __restrict__ W1T,
                             const float* __restrict__ b1,
                             unsigned short* __restrict__ H) {
  int lane = threadIdx.x & 31;
  int wid  = threadIdx.x >> 5;
  int mtile = blockIdx.x * 4 + wid;
  int nbase = blockIdx.y * 64;
  int z = blockIdx.z;
  int k = z % NBANDS;
  const unsigned short* A  = XN + (size_t)z * TN;
  const unsigned short* Bt = W1T + (size_t)k * (HID_DIM * FC_DIM);
  unsigned short* Out = H + (size_t)z * HN;
  v8f acc[4] = {{0.f,0.f,0.f,0.f,0.f,0.f,0.f,0.f},
                {0.f,0.f,0.f,0.f,0.f,0.f,0.f,0.f},
                {0.f,0.f,0.f,0.f,0.f,0.f,0.f,0.f},
                {0.f,0.f,0.f,0.f,0.f,0.f,0.f,0.f}};
  wmma_tile4(A, Bt, FC_DIM, FC_DIM, FC_DIM,
             mtile * 16, T_DIM - 1, nbase, HID_DIM, lane, acc);
#pragma unroll
  for (int j = 0; j < 4; ++j) {
    int ncol = nbase + j * 16 + (lane & 15);
    float bias = b1[k * HID_DIM + ncol];
    store_tile(acc[j], Out, HID_DIM, mtile * 16, T_DIM, ncol, HID_DIM,
               bias, true, lane);
  }
}

// generic batched GEMM (+bias), batch over blockIdx.z with element strides
__global__ void gemm_kernel(const unsigned short* __restrict__ A,
                            const unsigned short* __restrict__ Bt,
                            const float* __restrict__ bias,
                            unsigned short* __restrict__ Out,
                            int M, int Kd, int lda, int ldb, int ldout,
                            int Nreal, int Npad,
                            long long aStride, long long oStride) {
  int lane = threadIdx.x & 31;
  int wid  = threadIdx.x >> 5;
  int mtile = blockIdx.x * 4 + wid;
  int nbase = blockIdx.y * 64;
  if (nbase >= Npad) return;                         // wave-uniform
  const unsigned short* Ab = A + (size_t)blockIdx.z * aStride;
  unsigned short* Ob = Out + (size_t)blockIdx.z * oStride;
  v8f acc[4] = {{0.f,0.f,0.f,0.f,0.f,0.f,0.f,0.f},
                {0.f,0.f,0.f,0.f,0.f,0.f,0.f,0.f},
                {0.f,0.f,0.f,0.f,0.f,0.f,0.f,0.f},
                {0.f,0.f,0.f,0.f,0.f,0.f,0.f,0.f}};
  wmma_tile4(Ab, Bt, Kd, lda, ldb, mtile * 16, M - 1, nbase, Npad, lane, acc);
#pragma unroll
  for (int j = 0; j < 4; ++j) {
    int ncol = nbase + j * 16 + (lane & 15);
    float bv = (ncol < Nreal) ? bias[ncol] : 0.f;
    store_tile(acc[j], Ob, ldout, mtile * 16, M, ncol, Nreal, bv, false, lane);
  }
}

// GLU + faithful raw reshape/transpose scatter to complex output (float pairs)
__global__ void glu_scatter_kernel(const unsigned short* __restrict__ G,
                                   float* __restrict__ out,
                                   int d, int bw, int f0, long long gStride,
                                   int total /* = 517*d */) {
  int idx = blockIdx.x * 256 + threadIdx.x;
  int b = blockIdx.y;
  if (idx >= total) return;
  int t = idx / d, c = idx % d;
  const unsigned short* g = G + (size_t)b * gStride;
  float a  = bf2f(g[(size_t)t * (2 * d) + c]);
  float gg = bf2f(g[(size_t)t * (2 * d) + d + c]);
  float v  = a * (1.f / (1.f + expf(-gg)));
  int n = t * d + c;                 // flat index of o[b] as [T, 4*bw]
  int half = 2 * bw * T_DIM;
  int i = n / half;       int rem  = n % half;
  int j = rem / (bw * T_DIM); int rem2 = rem % (bw * T_DIM);
  int w = rem2 / T_DIM;   int u = rem2 % T_DIM;
  out[((((size_t)b * 2 + i) * 1025 + (f0 + w)) * T_DIM + u) * 2 + j] = v;
}

// ---------------- host ----------------
extern "C" void kernel_launch(void* const* d_in, const int* in_sizes, int n_in,
                              void* d_out, int out_size, void* d_ws, size_t ws_size,
                              hipStream_t stream) {
  (void)in_sizes; (void)out_size; (void)ws_size;
  static const int BW[NBANDS] = {
      5,5,4,5,5,4,5,5,4,5,
      12,11,12,11,12,12,11,12,11,12,12,11,
      23,24,23,23,23,24,23,23,
      46,47,46,47,46,47,46,47,
      92,93,96};

  int D_[NBANDS], DP_[NBANDS], NP_[NBANDS], N2P_[NBANDS];
  long long cBW[NBANDS + 1], cD[NBANDS + 1], cDP[NBANDS + 1], c2D[NBANDS + 1];
  long long cNP[NBANDS + 1], cWGT[NBANDS + 1], cWG[NBANDS + 1];
  cBW[0] = cD[0] = cDP[0] = c2D[0] = cNP[0] = cWGT[0] = cWG[0] = 0;
  for (int i = 0; i < NBANDS; ++i) {
    int d = 4 * BW[i];
    D_[i]   = d;
    DP_[i]  = ((d + 31) / 32) * 32;          // K-pad for GEMM3
    NP_[i]  = ((d + 15) / 16) * 16;          // N-pad for GEMM2
    N2P_[i] = ((2 * d + 15) / 16) * 16;      // N-pad for GEMM3
    cBW[i + 1] = cBW[i] + BW[i];
    cD [i + 1] = cD [i] + d;
    cDP[i + 1] = cDP[i] + DP_[i];
    c2D[i + 1] = c2D[i] + 2 * d;
    cNP[i + 1] = cNP[i] + NP_[i];
    cWGT[i + 1] = cWGT[i] + (long long)N2P_[i] * DP_[i];
    cWG [i + 1] = cWG [i] + (long long)d * 2 * d;
  }
  const long long sumDP = cDP[NBANDS];   // 4544
  const long long sum2D = c2D[NBANDS];   // 8200
  const long long sumNP = cNP[NBANDS];   // 4288

  // input pointers: handle expanded tuples (167 inputs) or concatenated (7)
  const float* x  = (const float*)d_in[0];
  const float* w1 = (const float*)d_in[1];
  const float* b1 = (const float*)d_in[2];
  const float *w2p[NBANDS], *b2p[NBANDS], *wgp[NBANDS], *bgp[NBANDS];
  if (n_in >= 3 + 4 * NBANDS) {
    for (int i = 0; i < NBANDS; ++i) {
      w2p[i] = (const float*)d_in[3 + i];
      b2p[i] = (const float*)d_in[3 + NBANDS + i];
      wgp[i] = (const float*)d_in[3 + 2 * NBANDS + i];
      bgp[i] = (const float*)d_in[3 + 3 * NBANDS + i];
    }
  } else {
    const float* w2c = (const float*)d_in[3];
    const float* b2c = (const float*)d_in[4];
    const float* wgc = (const float*)d_in[5];
    const float* bgc = (const float*)d_in[6];
    for (int i = 0; i < NBANDS; ++i) {
      w2p[i] = w2c + 512 * cD[i];
      b2p[i] = b2c + cD[i];
      wgp[i] = wgc + cWG[i];
      bgp[i] = bgc + c2D[i];
    }
  }

  // workspace layout (256B-aligned regions)
  char* ws = (char*)d_ws;
  size_t off = 0;
  auto alloc = [&](long long elems, int esz) {
    size_t o = off;
    off = (off + (size_t)elems * esz + 255) & ~(size_t)255;
    return o;
  };
  const long long NZ = 8LL * NBANDS;                 // 328
  unsigned short* XN  = (unsigned short*)(ws + alloc(NZ * TN, 2));
  unsigned short* H   = (unsigned short*)(ws + alloc(NZ * HN, 2));
  unsigned short* O   = (unsigned short*)(ws + alloc(8LL * T_DIM * sumDP, 2));
  unsigned short* G   = (unsigned short*)(ws + alloc(8LL * T_DIM * sum2D, 2));
  unsigned short* W1T = (unsigned short*)(ws + alloc((long long)NBANDS * HID_DIM * FC_DIM, 2));
  unsigned short* W2T = (unsigned short*)(ws + alloc(sumNP * 512, 2));
  unsigned short* WGT = (unsigned short*)(ws + alloc(cWGT[NBANDS], 2));
  float* out = (float*)d_out;

  // 1) LayerNorm -> bf16 xn
  layernorm_kernel<<<dim3((unsigned)NZ), dim3(256), 0, stream>>>(x, XN);

  // 2) pack weights (transpose to [N,K] bf16, zero-padded)
  for (int k = 0; k < NBANDS; ++k) {   // w1[k]: [128,512] -> [512,128]
    int tot = HID_DIM * FC_DIM;
    pack_bt_kernel<<<dim3((tot + 255) / 256), dim3(256), 0, stream>>>(
        w1 + (size_t)k * FC_DIM * HID_DIM, W1T + (size_t)k * HID_DIM * FC_DIM,
        HID_DIM, FC_DIM, HID_DIM, FC_DIM);
  }
  for (int i = 0; i < NBANDS; ++i) {   // w2[i]: [512,d] -> [NP,512]
    int tot = NP_[i] * 512;
    pack_bt_kernel<<<dim3((tot + 255) / 256), dim3(256), 0, stream>>>(
        w2p[i], W2T + cNP[i] * 512, D_[i], 512, NP_[i], 512);
  }
  for (int i = 0; i < NBANDS; ++i) {   // wg[i]: [d,2d] -> [N2P,DP]
    int tot = N2P_[i] * DP_[i];
    pack_bt_kernel<<<dim3((tot + 255) / 256), dim3(256), 0, stream>>>(
        wgp[i], WGT + cWGT[i], 2 * D_[i], D_[i], N2P_[i], DP_[i]);
  }

  // 3) zero o-buffer (K-dim zero padding for GEMM3)
  {
    long long nwords = 8LL * T_DIM * sumDP * 2 / 4;
    zero_u32_kernel<<<dim3((unsigned)((nwords + 255) / 256)), dim3(256), 0, stream>>>(
        (unsigned int*)O, nwords);
  }

  // 4) GEMM1 + tanh: h = tanh(xn @ w1 + b1), all (b,k) batched
  //    block = 4 waves over M (mtiles 0..35, clamped), y covers 64 cols each
  gemm1_kernel<<<dim3(9, HID_DIM / 64, (unsigned)NZ), dim3(128), 0, stream>>>(
      XN, W1T, b1, H);

  // 5) GEMM2 per band: o = h[:,i] @ w2[i] + b2[i]
  for (int i = 0; i < NBANDS; ++i) {
    unsigned gy = (unsigned)((NP_[i] + 63) / 64);
    gemm_kernel<<<dim3(9, gy, 8), dim3(128), 0, stream>>>(
        H + (size_t)i * HN, W2T + cNP[i] * 512, b2p[i],
        O + (size_t)T_DIM * cDP[i],
        T_DIM, 512, 512, 512, DP_[i], D_[i], NP_[i],
        (long long)NBANDS * HN, (long long)T_DIM * sumDP);
  }

  // 6) GEMM3 per band: g = o @ wg[i] + bg[i]
  for (int i = 0; i < NBANDS; ++i) {
    unsigned gy = (unsigned)((N2P_[i] + 63) / 64);
    gemm_kernel<<<dim3(9, gy, 8), dim3(128), 0, stream>>>(
        O + (size_t)T_DIM * cDP[i], WGT + cWGT[i], bgp[i],
        G + (size_t)T_DIM * c2D[i],
        T_DIM, DP_[i], DP_[i], DP_[i], 2 * D_[i], 2 * D_[i], N2P_[i],
        (long long)T_DIM * sumDP, (long long)T_DIM * sum2D);
  }

  // 7) GLU + scatter into complex output layout
  for (int i = 0; i < NBANDS; ++i) {
    int total = T_DIM * D_[i];
    glu_scatter_kernel<<<dim3((total + 255) / 256, 8), dim3(256), 0, stream>>>(
        G + (size_t)T_DIM * c2D[i], out, D_[i], BW[i], (int)cBW[i],
        (long long)T_DIM * sum2D, total);
  }
}
